// PCDet3DNet_22428319220765
// MI455X (gfx1250) — matrix operand order repaired and checked by reading, not compiled, not run on hardware
//
#include <hip/hip_runtime.h>
#include <hip/hip_bf16.h>

typedef __attribute__((ext_vector_type(16))) __bf16 v16bf;
typedef __attribute__((ext_vector_type(8)))  float  v8f;
typedef unsigned short u16b;
typedef __hip_bfloat16 bf16;

union Frag {
    v16bf  v;
    u16b   u[16];
    float4 q[2];
};

// ---------------------------------------------------------------------------
// Scatter points into dense bf16 grid + occupancy mask (grid pre-zeroed).
// ---------------------------------------------------------------------------
__global__ void scatter_kernel(const float* __restrict__ x,
                               const int* __restrict__ coords,
                               bf16* __restrict__ feat,
                               unsigned char* __restrict__ mask,
                               int N_, int H, int W, int C)
{
    int i = blockIdx.x * blockDim.x + threadIdx.x;
    if (i >= N_) return;
    int z  = coords[i * 4 + 1];
    int y  = coords[i * 4 + 2];
    int xx = coords[i * 4 + 3];
    size_t vox = ((size_t)z * H + y) * W + xx;
    for (int c = 0; c < C; ++c)
        feat[vox * C + c] = bf16(x[i * C + c]);
    mask[vox] = 1;
}

// ---------------------------------------------------------------------------
// Mask dilation + downsample for strided ('sp') layers.
// ---------------------------------------------------------------------------
__global__ void mask_down_kernel(const unsigned char* __restrict__ min_,
                                 unsigned char* __restrict__ mout,
                                 int Din, int Hin, int Win,
                                 int Dout, int Hout, int Wout,
                                 int kd, int kh, int kw,
                                 int sz, int sy, int sx,
                                 int pz, int py, int px)
{
    unsigned M = (unsigned)(Dout * Hout * Wout);
    unsigned i = blockIdx.x * blockDim.x + threadIdx.x;
    if (i >= M) return;
    unsigned t = i;
    int xo = (int)(t % (unsigned)Wout); t /= (unsigned)Wout;
    int yo = (int)(t % (unsigned)Hout);
    int zo = (int)(t / (unsigned)Hout);
    unsigned char m = 0;
    for (int dz = 0; dz < kd && !m; ++dz) {
        int zi = zo * sz - pz + dz;
        if ((unsigned)zi >= (unsigned)Din) continue;
        for (int dy = 0; dy < kh && !m; ++dy) {
            int yi = yo * sy - py + dy;
            if ((unsigned)yi >= (unsigned)Hin) continue;
            for (int dx = 0; dx < kw; ++dx) {
                int xi = xo * sx - px + dx;
                if ((unsigned)xi < (unsigned)Win &&
                    min_[((size_t)zi * Hin + yi) * Win + xi]) { m = 1; break; }
            }
        }
    }
    mout[i] = m;
}

// ---------------------------------------------------------------------------
// Fold BN into per-channel scale/bias.
// ---------------------------------------------------------------------------
__global__ void bn_prep_kernel(const float* __restrict__ gamma,
                               const float* __restrict__ beta,
                               const float* __restrict__ mean,
                               const float* __restrict__ var,
                               float* __restrict__ scale,
                               float* __restrict__ bias, int C)
{
    int i = blockIdx.x * blockDim.x + threadIdx.x;
    if (i >= C) return;
    float s = gamma[i] * rsqrtf(var[i] + 1e-3f);
    scale[i] = s;
    bias[i]  = beta[i] - mean[i] * s;
}

// ---------------------------------------------------------------------------
// Repack f32 weights (kd,kh,kw,cin,cout) into bf16 B-fragment order:
// dst[kchunk][ntile][lane][elem]; lane j<16: K = kc*32+{0-7,16-23}, lane>=16: {8-15,24-31}.
// ---------------------------------------------------------------------------
__global__ void pack_w_kernel(const float* __restrict__ W,
                              bf16* __restrict__ dst,
                              int ntaps, int Cin, int Cout)
{
    int Ktot = ntaps * Cin;
    int KC   = (Ktot + 31) >> 5;
    int NT   = Cout >> 4;
    long total = (long)KC * NT * 512;
    long i = (long)blockIdx.x * blockDim.x + threadIdx.x;
    if (i >= total) return;
    int  e    = (int)(i & 15);
    long t    = i >> 4;
    int  lane = (int)(t & 31); t >>= 5;
    int  nt   = (int)(t % NT);
    int  kc   = (int)(t / NT);
    int  K    = kc * 32 + ((lane >> 4) << 3) + (e < 8 ? e : e + 8);
    int  n    = nt * 16 + (lane & 15);
    float w = 0.f;
    if (K < Ktot) {
        int tap = K / Cin, c = K % Cin;
        w = W[((size_t)tap * Cin + c) * Cout + n];
    }
    dst[i] = bf16(w);
}

// ---------------------------------------------------------------------------
// Implicit-GEMM 3D conv, fully specialized on channel/kernel geometry so all
// inner-loop integer div/mod are by compile-time constants. One wave32
// computes a 16-voxel x (16*NPW)-cout tile: each gathered A fragment feeds
// NPW v_wmma_f32_16x16x32_bf16 ops. Packed weights are staged into LDS once
// per block (when they fit in 128KB) and served via ds_load_b128; larger
// weight sets stay on the global/L2 path. Fused BN+ReLU+mask epilogue.
// ---------------------------------------------------------------------------
template<int CIN, int COUT, int KD, int KH, int KW, bool F32OUT>
__global__ void conv_wmma_kernel(const bf16* __restrict__ fin,
                                 int Din, int Hin, int Win,
                                 void* __restrict__ fout,
                                 int Dout, int Hout, int Wout,
                                 int sz, int sy, int sx,
                                 int pz, int py, int px,
                                 const bf16* __restrict__ wpk,
                                 const float* __restrict__ scale,
                                 const float* __restrict__ bias,
                                 const unsigned char* __restrict__ mask)
{
    constexpr int NT    = COUT / 16;
    constexpr int NPW   = (NT >= 4) ? 4 : NT;  // ntiles per wave (A reuse)
    constexpr int NG    = NT / NPW;            // ntile groups
    constexpr int NTAPS = KD * KH * KW;
    constexpr int KTOT  = NTAPS * CIN;
    constexpr int KC    = (KTOT + 31) / 32;
    constexpr unsigned WBYTES = (unsigned)KC * NT * 1024u;  // packed wt bytes
    constexpr bool USELDS = (WBYTES <= 128u * 1024u);

    extern __shared__ __align__(16) char smem_raw[];

    // Stage packed weights in LDS (all waves participate, before any exit).
    if constexpr (USELDS) {
        const uint4* src = (const uint4*)wpk;
        uint4*       dst = (uint4*)smem_raw;
        constexpr int NV = (int)(WBYTES / 16);
        for (int i = (int)threadIdx.x; i < NV; i += 256) dst[i] = src[i];
        __syncthreads();
    }

    const int lane   = threadIdx.x & 31;
    const int wave   = threadIdx.x >> 5;
    const int laneHi = lane >> 4;

    const unsigned M      = (unsigned)(Dout * Hout * Wout);
    const unsigned Mtiles = (M + 15u) >> 4;
    unsigned gw = blockIdx.x * (blockDim.x >> 5) + wave;
    if (gw >= Mtiles * NG) return;         // wave-uniform: EXEC stays all-1s

    const unsigned mtile  = gw / NG;       // NG is compile-time
    const int      ntile0 = (int)(gw % NG) * NPW;

    // This lane's A-matrix row -> output voxel coordinates.
    const unsigned mvox   = mtile * 16u + (lane & 15);
    const bool     mvalid = (mvox < M);
    int zo = 0, yo = 0, xo = 0;
    if (mvalid) {
        unsigned t = mvox;
        xo = (int)(t % (unsigned)Wout); t /= (unsigned)Wout;
        yo = (int)(t % (unsigned)Hout);
        zo = (int)(t / (unsigned)Hout);
    }

    v8f acc[NPW];
#pragma unroll
    for (int np = 0; np < NPW; ++np) acc[np] = (v8f){};

    for (int kc = 0; kc < KC; ++kc) {
        Frag a;
        // A fragment gather. ISA lane layout: lane<16 holds K {0-7,16-23},
        // lane>=16 holds {8-15,24-31} of the chunk; runs of 8 consecutive K.
        if constexpr (CIN >= 8) {
#pragma unroll
            for (int r = 0; r < 2; ++r) {
                const int K0  = kc * 32 + laneHi * 8 + r * 16;
                const int tap = K0 / CIN;       // constant divisor
                const int c0  = K0 % CIN;
                float4 val = {0.f, 0.f, 0.f, 0.f};
                if (mvalid && tap < NTAPS) {
                    const int dz = tap / (KH * KW), rem = tap % (KH * KW);
                    const int dy = rem / KW,        dx  = rem % KW;
                    const int zi = zo * sz - pz + dz;
                    const int yi = yo * sy - py + dy;
                    const int xi = xo * sx - px + dx;
                    if ((unsigned)zi < (unsigned)Din &&
                        (unsigned)yi < (unsigned)Hin &&
                        (unsigned)xi < (unsigned)Win) {
                        size_t o = (((size_t)zi * Hin + yi) * Win + xi) * CIN + c0;
                        val = *(const float4*)(fin + o);
                    }
                }
                a.q[r] = val;
            }
        } else {  // CIN = 4 (layer 0): elementwise d16 gathers
#pragma unroll
            for (int e = 0; e < 16; ++e) {
                const int K = kc * 32 + laneHi * 8 + (e < 8 ? e : 8 + e);
                u16b val = 0;
                if (mvalid && K < KTOT) {
                    const int tap = K / CIN, c = K % CIN;
                    const int dz = tap / (KH * KW), rem = tap % (KH * KW);
                    const int dy = rem / KW,        dx  = rem % KW;
                    const int zi = zo * sz - pz + dz;
                    const int yi = yo * sy - py + dy;
                    const int xi = xo * sx - px + dx;
                    if ((unsigned)zi < (unsigned)Din &&
                        (unsigned)yi < (unsigned)Hin &&
                        (unsigned)xi < (unsigned)Win) {
                        size_t o = (((size_t)zi * Hin + yi) * Win + xi) * CIN + c;
                        val = *(const u16b*)(fin + o);
                    }
                }
                a.u[e] = val;
            }
        }

        // B fragments: from LDS (ds_load_b128) or global (L2-resident).
        const size_t boff = ((size_t)(kc * NT + ntile0) * 32 + lane) * 16;
        Frag b[NPW];
        if constexpr (USELDS) {
            const bf16* bpb = (const bf16*)smem_raw + boff;
#pragma unroll
            for (int np = 0; np < NPW; ++np) {
                b[np].q[0] = *(const float4*)(bpb + (size_t)np * 512);
                b[np].q[1] = *(const float4*)(bpb + (size_t)np * 512 + 8);
            }
        } else {
            const bf16* bpb = wpk + boff;
#pragma unroll
            for (int np = 0; np < NPW; ++np) {
                b[np].q[0] = *(const float4*)(bpb + (size_t)np * 512);
                b[np].q[1] = *(const float4*)(bpb + (size_t)np * 512 + 8);
            }
            if (kc + 1 < KC)
                __builtin_prefetch((const void*)(bpb + (size_t)NT * 512), 0, 0);
        }
#pragma unroll
        for (int np = 0; np < NPW; ++np)
            acc[np] = __builtin_amdgcn_wmma_f32_16x16x32_bf16(
                false, a.v, false, b[np].v, (short)0, acc[np], false, false);
    }

    // Epilogue: C layout -> VGPR v holds (M = v + 8*laneHi, N = lane&15).
    float sc[NPW], bi[NPW];
#pragma unroll
    for (int np = 0; np < NPW; ++np) {
        const int n = (ntile0 + np) * 16 + (lane & 15);
        sc[np] = scale[n];
        bi[np] = bias[n];
    }
#pragma unroll
    for (int v = 0; v < 8; ++v) {
        const unsigned mm = mtile * 16u + v + laneHi * 8;
        if (mm >= M) continue;
        const float mval = (float)mask[mm];
#pragma unroll
        for (int np = 0; np < NPW; ++np) {
            const int n = (ntile0 + np) * 16 + (lane & 15);
            float val = acc[np][v] * sc[np] + bi[np];
            val = val > 0.f ? val : 0.f;
            val *= mval;
            if constexpr (F32OUT) ((float*)fout)[(size_t)mm * COUT + n] = val;
            else                  ((bf16*) fout)[(size_t)mm * COUT + n] = bf16(val);
        }
    }
}

// ---------------------------------------------------------------------------
// Host side.
// ---------------------------------------------------------------------------
static const int RD[5][3] = {{41,400,352},{21,200,176},{11,100,88},{5,50,44},{3,50,44}};

template<int CIN, int COUT, int KD, int KH, int KW, bool F32OUT>
static void launch_conv(const bf16* fin, const int* rin, void* fout, const int* rout,
                        int sz, int sy, int sx, int pz, int py, int px,
                        const bf16* wpk, const float* sc, const float* bi,
                        const unsigned char* mask, hipStream_t stream)
{
    constexpr int NT = COUT / 16, NPW = (NT >= 4) ? 4 : NT, NG = NT / NPW;
    constexpr int NTAPS = KD * KH * KW;
    constexpr int KC    = (NTAPS * CIN + 31) / 32;
    constexpr unsigned WBYTES = (unsigned)KC * NT * 1024u;
    constexpr unsigned SHMEM  = (WBYTES <= 128u * 1024u) ? WBYTES : 0u;
    long M     = (long)rout[0] * rout[1] * rout[2];
    long tiles = ((M + 15) / 16) * NG;
    long blks  = (tiles + 7) / 8;           // 8 waves (256 threads) per block
    conv_wmma_kernel<CIN, COUT, KD, KH, KW, F32OUT>
        <<<dim3((unsigned)blks), dim3(256), SHMEM, stream>>>(
            fin, rin[0], rin[1], rin[2], fout, rout[0], rout[1], rout[2],
            sz, sy, sx, pz, py, px, wpk, sc, bi, mask);
}

struct LSpec { int cin, cout, kd, kh, kw, sz, sy, sx, pz, py, px, rin, rout; };
static const LSpec Ls[12] = {
    { 4, 16, 3,3,3, 1,1,1, 1,1,1, 0,0},
    {16, 16, 3,3,3, 1,1,1, 1,1,1, 0,0},
    {16, 32, 3,3,3, 2,2,2, 1,1,1, 0,1},
    {32, 32, 3,3,3, 1,1,1, 1,1,1, 1,1},
    {32, 32, 3,3,3, 1,1,1, 1,1,1, 1,1},
    {32, 64, 3,3,3, 2,2,2, 1,1,1, 1,2},
    {64, 64, 3,3,3, 1,1,1, 1,1,1, 2,2},
    {64, 64, 3,3,3, 1,1,1, 1,1,1, 2,2},
    {64, 64, 3,3,3, 2,2,2, 0,1,1, 2,3},
    {64, 64, 3,3,3, 1,1,1, 1,1,1, 3,3},
    {64, 64, 3,3,3, 1,1,1, 1,1,1, 3,3},
    {64,128, 3,1,1, 2,1,1, 1,0,0, 3,4},
};

extern "C" void kernel_launch(void* const* d_in, const int* in_sizes, int n_in,
                              void* d_out, int out_size, void* d_ws, size_t ws_size,
                              hipStream_t stream)
{
    (void)in_sizes; (void)n_in; (void)out_size; (void)ws_size;
    const float* x      = (const float*)d_in[0];
    const int*   coords = (const int*)d_in[1];
    // d_in[2] = points_mean (unused by the reference forward pass)
    // d_in[3 + L*5 + {0..4}] = {W, gamma, beta, mean, var} per layer

    char* ws = (char*)d_ws;
    size_t off = 0;
    auto alloc = [&](size_t bytes) -> char* {
        char* p = ws + off;
        off = (off + bytes + 255) & ~(size_t)255;
        return p;
    };

    const long Rv[5] = {
        (long)41*400*352, (long)21*200*176, (long)11*100*88,
        (long)5*50*44,    (long)3*50*44
    };

    // Feature grids (bf16) and masks (u8)
    bf16* g0a = (bf16*)alloc(Rv[0]*4*2);
    bf16* g0b = (bf16*)alloc(Rv[0]*16*2);
    bf16* g0c = (bf16*)alloc(Rv[0]*16*2);
    bf16* g1a = (bf16*)alloc(Rv[1]*32*2);
    bf16* g1b = (bf16*)alloc(Rv[1]*32*2);
    bf16* g2a = (bf16*)alloc(Rv[2]*64*2);
    bf16* g2b = (bf16*)alloc(Rv[2]*64*2);
    bf16* g3a = (bf16*)alloc(Rv[3]*64*2);
    bf16* g3b = (bf16*)alloc(Rv[3]*64*2);
    unsigned char* mk[5];
    for (int r = 0; r < 5; ++r) mk[r] = (unsigned char*)alloc((size_t)Rv[r]);

    // Packed weights + folded BN per layer
    bf16*  wpk[12];
    float* bns[12];
    float* bnb[12];
    for (int L = 0; L < 12; ++L) {
        int ntaps = Ls[L].kd * Ls[L].kh * Ls[L].kw;
        int Ktot  = ntaps * Ls[L].cin;
        int KC    = (Ktot + 31) >> 5;
        int NT    = Ls[L].cout >> 4;
        wpk[L] = (bf16*)alloc((size_t)KC * NT * 512 * 2);
        bns[L] = (float*)alloc((size_t)Ls[L].cout * 4);
        bnb[L] = (float*)alloc((size_t)Ls[L].cout * 4);
    }

    // --- init + scatter ---
    hipMemsetAsync(g0a,   0, (size_t)Rv[0]*4*2, stream);
    hipMemsetAsync(mk[0], 0, (size_t)Rv[0],     stream);
    scatter_kernel<<<dim3((40000 + 255) / 256), dim3(256), 0, stream>>>(
        x, coords, g0a, mk[0], 40000, RD[0][1], RD[0][2], 4);

    // --- weight / BN prep ---
    for (int L = 0; L < 12; ++L) {
        const LSpec& s = Ls[L];
        const float* W     = (const float*)d_in[3 + L*5 + 0];
        const float* gamma = (const float*)d_in[3 + L*5 + 1];
        const float* beta  = (const float*)d_in[3 + L*5 + 2];
        const float* mean  = (const float*)d_in[3 + L*5 + 3];
        const float* var   = (const float*)d_in[3 + L*5 + 4];
        bn_prep_kernel<<<dim3((s.cout + 63) / 64), dim3(64), 0, stream>>>(
            gamma, beta, mean, var, bns[L], bnb[L], s.cout);
        int ntaps = s.kd * s.kh * s.kw;
        int Ktot  = ntaps * s.cin;
        int KC    = (Ktot + 31) >> 5;
        int NT    = s.cout >> 4;
        long tot  = (long)KC * NT * 512;
        pack_w_kernel<<<dim3((unsigned)((tot + 255) / 256)), dim3(256), 0, stream>>>(
            W, wpk[L], ntaps, s.cin, s.cout);
    }

    // --- mask pyramid (strided layers 2, 5, 8, 11) ---
    const int mlayers[4] = {2, 5, 8, 11};
    for (int i = 0; i < 4; ++i) {
        const LSpec& s = Ls[mlayers[i]];
        long M = Rv[s.rout];
        mask_down_kernel<<<dim3((unsigned)((M + 255) / 256)), dim3(256), 0, stream>>>(
            mk[s.rin], mk[s.rout],
            RD[s.rin][0], RD[s.rin][1], RD[s.rin][2],
            RD[s.rout][0], RD[s.rout][1], RD[s.rout][2],
            s.kd, s.kh, s.kw, s.sz, s.sy, s.sx, s.pz, s.py, s.px);
    }

    // --- conv chain (explicit instantiations per layer geometry) ---
    launch_conv< 4, 16,3,3,3,false>(g0a, RD[0], g0b, RD[0], 1,1,1, 1,1,1, wpk[0],  bns[0],  bnb[0],  mk[0], stream);
    launch_conv<16, 16,3,3,3,false>(g0b, RD[0], g0c, RD[0], 1,1,1, 1,1,1, wpk[1],  bns[1],  bnb[1],  mk[0], stream);
    launch_conv<16, 32,3,3,3,false>(g0c, RD[0], g1a, RD[1], 2,2,2, 1,1,1, wpk[2],  bns[2],  bnb[2],  mk[1], stream);
    launch_conv<32, 32,3,3,3,false>(g1a, RD[1], g1b, RD[1], 1,1,1, 1,1,1, wpk[3],  bns[3],  bnb[3],  mk[1], stream);
    launch_conv<32, 32,3,3,3,false>(g1b, RD[1], g1a, RD[1], 1,1,1, 1,1,1, wpk[4],  bns[4],  bnb[4],  mk[1], stream);
    launch_conv<32, 64,3,3,3,false>(g1a, RD[1], g2a, RD[2], 2,2,2, 1,1,1, wpk[5],  bns[5],  bnb[5],  mk[2], stream);
    launch_conv<64, 64,3,3,3,false>(g2a, RD[2], g2b, RD[2], 1,1,1, 1,1,1, wpk[6],  bns[6],  bnb[6],  mk[2], stream);
    launch_conv<64, 64,3,3,3,false>(g2b, RD[2], g2a, RD[2], 1,1,1, 1,1,1, wpk[7],  bns[7],  bnb[7],  mk[2], stream);
    launch_conv<64, 64,3,3,3,false>(g2a, RD[2], g3a, RD[3], 2,2,2, 0,1,1, wpk[8],  bns[8],  bnb[8],  mk[3], stream);
    launch_conv<64, 64,3,3,3,false>(g3a, RD[3], g3b, RD[3], 1,1,1, 1,1,1, wpk[9],  bns[9],  bnb[9],  mk[3], stream);
    launch_conv<64, 64,3,3,3,false>(g3b, RD[3], g3a, RD[3], 1,1,1, 1,1,1, wpk[10], bns[10], bnb[10], mk[3], stream);
    launch_conv<64,128,3,1,1,true >(g3a, RD[3], d_out, RD[4], 2,1,1, 1,0,0, wpk[11], bns[11], bnb[11], mk[4], stream);
}
